// MultiHeadedAttention_76682346103448
// MI455X (gfx1250) — matrix operand order repaired and checked by reading, not compiled, not run on hardware
//
#include <hip/hip_runtime.h>

#define DFEAT 1024
#define NHEAD 16
#define DHEAD 64
#define BATCH 2
#define SEQ   2048
#define MROWS (BATCH * SEQ)   // 4096

typedef _Float16 half_t;
typedef __attribute__((ext_vector_type(16))) _Float16 v16h;
typedef __attribute__((ext_vector_type(8)))  _Float16 v8h;
typedef __attribute__((ext_vector_type(8)))  float    v8f;
typedef __attribute__((ext_vector_type(4)))  float    v4f;
typedef __attribute__((ext_vector_type(4)))  unsigned u32x4;
typedef __attribute__((ext_vector_type(8)))  unsigned u32x8;

// Build a 16-half WMMA fragment from two contiguous 8-half (16B) chunks.
static __device__ __forceinline__ v16h frag16(const half_t* p0, const half_t* p1) {
  v8h lo = *(const v8h*)p0;
  v8h hi = *(const v8h*)p1;
  v16h r;
#pragma unroll
  for (int i = 0; i < 8; ++i) { r[i] = lo[i]; r[i + 8] = hi[i]; }
  return r;
}

static __device__ __forceinline__ v8f wmma_f16(v16h a, v16h b, v8f c) {
  return __builtin_amdgcn_wmma_f32_16x16x32_f16(false, a, false, b, (short)0, c,
                                                false, false);
}

// Wave32 XOR-lane shuffle as a single ds_swizzle_b32 with immediate pattern.
template <int MASK>
static __device__ __forceinline__ float swz_xor(float v) {
  int r = __builtin_amdgcn_ds_swizzle(__builtin_bit_cast(int, v),
                                      (MASK << 10) | 0x1f);
  return __builtin_bit_cast(float, r);
}

static __device__ __forceinline__ float red_max16(float v) {
  v = fmaxf(v, swz_xor<1>(v));
  v = fmaxf(v, swz_xor<2>(v));
  v = fmaxf(v, swz_xor<4>(v));
  v = fmaxf(v, swz_xor<8>(v));
  return v;
}
static __device__ __forceinline__ float red_sum16(float v) {
  v += swz_xor<1>(v);
  v += swz_xor<2>(v);
  v += swz_xor<4>(v);
  v += swz_xor<8>(v);
  return v;
}

// LDS byte offset of a __shared__ object (flat LDS aperture keeps it in [31:0]).
static __device__ __forceinline__ unsigned lds_off(const void* p) {
  return (unsigned)(unsigned long long)p;
}

// Issue one 2D TDM tile move Global->LDS: tile0 x tile1 elements of 2 bytes,
// line stride `stride0` elements. Portable inline-asm form.
static __device__ __forceinline__ void tdm_load_2d_f16(unsigned lds_addr,
                                                       const half_t* gptr,
                                                       unsigned tile0,
                                                       unsigned tile1,
                                                       unsigned stride0) {
  unsigned long long ga = (unsigned long long)gptr;
  u32x4 g0;
  g0.x = 1u;                                    // count=1 valid descriptor
  g0.y = lds_addr;                              // lds_addr (bytes)
  g0.z = (unsigned)ga;                          // global_addr[31:0]
  g0.w = (unsigned)(ga >> 32) | 0x80000000u;    // global_addr[56:32] | type=2
  u32x8 g1;
  g1.s0 = 0x00010000u;                          // wg_mask=0, data_size=1 (2B)
  g1.s1 = tile0 << 16;                          // tensor_dim0[15:0]
  g1.s2 = tile1 << 16;                          // tensor_dim0 hi | tensor_dim1 lo
  g1.s3 = tile0 << 16;                          // tensor_dim1 hi | tile_dim0
  g1.s4 = tile1;                                // tile_dim1 | tile_dim2=0
  g1.s5 = stride0;                              // tensor_dim0_stride[31:0]
  g1.s6 = 0u;
  g1.s7 = 0u;
  asm volatile("tensor_load_to_lds %0, %1" :: "s"(g0), "s"(g1) : "memory");
}

// ------------------------------------------------- f32 -> f16, 8 elems/thread
__global__ void cvt_f32_to_f16(const float* __restrict__ src,
                               half_t* __restrict__ dst, int n8) {
  int i = blockIdx.x * blockDim.x + threadIdx.x;
  int stride = gridDim.x * blockDim.x;
  for (; i < n8; i += stride) {
    v4f a = *(const v4f*)(src + (size_t)i * 8);
    v4f b = *(const v4f*)(src + (size_t)i * 8 + 4);
    v8h o;
#pragma unroll
    for (int k = 0; k < 4; ++k) { o[k] = (half_t)a[k]; o[k + 4] = (half_t)b[k]; }
    *(v8h*)(dst + (size_t)i * 8) = o;
  }
}

// ------------------------------------------------- C = A(f16) * W(f16)^T + b
// MODE 0: f16 row-major [MROWS][DFEAT]
// MODE 1: f16 per-head transposed [B][H][DHEAD][SEQ]  (for V)
// MODE 2: f32 row-major (final output)
// Block = 128 threads (4 waves). Wave computes a 16x64 output tile.
template <int MODE>
__global__ __launch_bounds__(128) void gemm_bias(const half_t* __restrict__ A,
                                                 const half_t* __restrict__ W,
                                                 const float* __restrict__ bias,
                                                 half_t* __restrict__ outH,
                                                 float* __restrict__ outF) {
  const int lane = threadIdx.x & 31;
  const int wave = threadIdx.x >> 5;
  const int hs = lane >> 4;
  const int ln = lane & 15;
  const int m0 = blockIdx.y * 64 + wave * 16;
  const int nb = blockIdx.x * 64;

  v8f zero = {};
  v8f acc[4];
#pragma unroll
  for (int t = 0; t < 4; ++t) acc[t] = zero;

  const half_t* arow = A + (size_t)(m0 + ln) * DFEAT;
#pragma unroll 2
  for (int k0 = 0; k0 < DFEAT; k0 += 32) {
    if (k0 + 32 < DFEAT) __builtin_prefetch(arow + k0 + 32, 0, 0);
    v16h a = frag16(arow + k0 + hs * 8, arow + k0 + 16 + hs * 8);
#pragma unroll
    for (int t = 0; t < 4; ++t) {
      const half_t* wp = W + (size_t)(nb + t * 16 + ln) * DFEAT + k0 + hs * 8;
      v16h b = frag16(wp, wp + 16);
      acc[t] = wmma_f16(a, b, acc[t]);
    }
  }
#pragma unroll
  for (int t = 0; t < 4; ++t) {
    int col = nb + t * 16 + ln;
    float bv = bias[col];
#pragma unroll
    for (int v = 0; v < 8; ++v) {
      int row = m0 + v + hs * 8;   // C layout: VGPR v -> row v + hs*8
      float val = acc[t][v] + bv;
      if (MODE == 2) {
        outF[(size_t)row * DFEAT + col] = val;
      } else if (MODE == 1) {
        int bb = row >> 11, s = row & (SEQ - 1);
        int hh = col >> 6,  d = col & (DHEAD - 1);
        outH[((size_t)((bb * NHEAD + hh) * DHEAD + d)) * SEQ + s] = (half_t)val;
      } else {
        outH[(size_t)row * DFEAT + col] = (half_t)val;
      }
    }
  }
}

// -------------------------------------------- fused flash-style attention
// Grid: (SEQ/64, BATCH*NHEAD), block 128. Wave w owns 16 query rows.
// K (row-major) and V (pre-transposed per head) 32-key tiles are DMA'd into
// double-buffered LDS by the Tensor Data Mover; wave 0 issues tile j+1 before
// waiting TENSORcnt<=2 so the DMA overlaps compute on tile j.
__global__ __launch_bounds__(128) void attn_fused(const half_t* __restrict__ Qf,
                                                  const half_t* __restrict__ Kf,
                                                  const half_t* __restrict__ Vt,
                                                  const int* __restrict__ mask,
                                                  half_t* __restrict__ Ctx) {
  __shared__ half_t Kl[2][32 * DHEAD];     // 2 x 4 KB, [key][d]
  __shared__ half_t Vl[2][DHEAD * 32];     // 2 x 4 KB, [d][key]
  __shared__ half_t Pl[4][16 * 32];        // 4 KB (per-wave P tiles)

  const int lane = threadIdx.x & 31;
  const int wave = threadIdx.x >> 5;
  const int hs = lane >> 4;
  const int ln = lane & 15;
  const int bh = blockIdx.y;
  const int b = bh >> 4;
  const int h = bh & 15;
  const int q0 = blockIdx.x * 64 + wave * 16;

  const half_t* qrow = Qf + (size_t)(b * SEQ + q0 + ln) * DFEAT + h * DHEAD;
  v16h aq0 = frag16(qrow + hs * 8,      qrow + 16 + hs * 8);
  v16h aq1 = frag16(qrow + 32 + hs * 8, qrow + 48 + hs * 8);

  const half_t* kbase = Kf + (size_t)(b * SEQ) * DFEAT + h * DHEAD;  // [s][d]
  const half_t* vbase = Vt + (size_t)(b * NHEAD + h) * DHEAD * SEQ;  // [d][s]

  // preload tile 0 into buffer 0
  if (wave == 0) {
    tdm_load_2d_f16(lds_off(&Kl[0][0]), kbase, DHEAD, 32, DFEAT);
    tdm_load_2d_f16(lds_off(&Vl[0][0]), vbase, 32, DHEAD, SEQ);
  }

  v8f zero = {};
  v8f o[4];
#pragma unroll
  for (int t = 0; t < 4; ++t) o[t] = zero;
  float mrun[8], lrun[8];
#pragma unroll
  for (int v = 0; v < 8; ++v) { mrun[v] = -1e30f; lrun[v] = 0.f; }

  const float scale = 0.03125f;   // 1/sqrt(D_FEAT) = 1/32

  for (int j = 0, it = 0; j < SEQ; j += 32, ++it) {
    const int buf = it & 1;
    if (wave == 0) {
      if (j + 32 < SEQ) {
        // async prefetch of next tile into the other buffer, then require the
        // current tile's two descriptors complete (TDM completes in order)
        tdm_load_2d_f16(lds_off(&Kl[buf ^ 1][0]), kbase + (size_t)(j + 32) * DFEAT,
                        DHEAD, 32, DFEAT);
        tdm_load_2d_f16(lds_off(&Vl[buf ^ 1][0]), vbase + (j + 32),
                        32, DHEAD, SEQ);
        __builtin_amdgcn_s_wait_tensorcnt(2);
      } else {
        __builtin_amdgcn_s_wait_tensorcnt(0);
      }
    }
    __syncthreads();   // publish current tile to all waves

    // ---- scores: S = Q(16x64) . K_blk^T  -> two 16x16 f32 tiles
    v8f s0 = zero, s1 = zero;
#pragma unroll
    for (int ks = 0; ks < 2; ++ks) {
      v16h a = ks ? aq1 : aq0;
      const half_t* kp0 = &Kl[buf][(ln)      * DHEAD + ks * 32 + hs * 8];
      const half_t* kp1 = &Kl[buf][(16 + ln) * DHEAD + ks * 32 + hs * 8];
      v16h b0 = frag16(kp0, kp0 + 16);
      v16h b1 = frag16(kp1, kp1 + 16);
      s0 = wmma_f16(a, b0, s0);
      s1 = wmma_f16(a, b1, s1);
    }

    // ---- scale + key mask
    int mk0 = mask[b * SEQ + j + ln];
    int mk1 = mask[b * SEQ + j + 16 + ln];
#pragma unroll
    for (int v = 0; v < 8; ++v) {
      s0[v] = mk0 ? s0[v] * scale : -1e9f;
      s1[v] = mk1 ? s1[v] * scale : -1e9f;
    }

    // ---- online softmax (row lives in one VGPR across a 16-lane half-wave)
    float corr[8];
#pragma unroll
    for (int v = 0; v < 8; ++v) {
      float mv = red_max16(fmaxf(s0[v], s1[v]));
      float mnew = fmaxf(mrun[v], mv);
      corr[v] = __expf(mrun[v] - mnew);
      s0[v] = __expf(s0[v] - mnew);
      s1[v] = __expf(s1[v] - mnew);
      float rs = red_sum16(s0[v] + s1[v]);
      lrun[v] = lrun[v] * corr[v] + rs;
      mrun[v] = mnew;
    }
#pragma unroll
    for (int t = 0; t < 4; ++t)
#pragma unroll
      for (int v = 0; v < 8; ++v) o[t][v] *= corr[v];

    // ---- transpose P (C-layout -> A-layout) through per-wave LDS tile
    half_t* P = &Pl[wave][0];
#pragma unroll
    for (int v = 0; v < 8; ++v) {
      P[(hs * 8 + v) * 32 + ln]      = (half_t)s0[v];
      P[(hs * 8 + v) * 32 + 16 + ln] = (half_t)s1[v];
    }
    __syncthreads();   // orders DS stores before DS reads below

    const half_t* pr = &P[ln * 32 + hs * 8];
    v16h pa = frag16(pr, pr + 16);

    // ---- O += P(16x32) . V_blk(32x64): V is [d][key] in LDS, so each
    // B-fragment is two contiguous ds_load_b128 per lane.
#pragma unroll
    for (int t = 0; t < 4; ++t) {
      const half_t* vp = &Vl[buf][(t * 16 + ln) * 32 + hs * 8];
      v16h bvf = frag16(vp, vp + 16);
      o[t] = wmma_f16(pa, bvf, o[t]);
    }
    __syncthreads();   // all waves done with this buffer before TDM refills it
  }

  // ---- epilogue: normalize and pack back to [B,S,DFEAT] f16 context
#pragma unroll
  for (int t = 0; t < 4; ++t) {
#pragma unroll
    for (int v = 0; v < 8; ++v) {
      int row = q0 + v + hs * 8;
      int col = h * DHEAD + t * 16 + ln;
      Ctx[(size_t)(b * SEQ + row) * DFEAT + col] = (half_t)(o[t][v] / lrun[v]);
    }
  }
}

// ---------------------------------------------------------------------------
extern "C" void kernel_launch(void* const* d_in, const int* in_sizes, int n_in,
                              void* d_out, int out_size, void* d_ws, size_t ws_size,
                              hipStream_t stream) {
  (void)in_sizes; (void)n_in; (void)out_size; (void)ws_size;
  const float* Q   = (const float*)d_in[0];
  const float* K   = (const float*)d_in[1];
  const float* V   = (const float*)d_in[2];
  const int*   msk = (const int*)  d_in[3];
  const float* Wq  = (const float*)d_in[4];
  const float* bq  = (const float*)d_in[5];
  const float* Wk  = (const float*)d_in[6];
  const float* bk  = (const float*)d_in[7];
  const float* Wv  = (const float*)d_in[8];
  const float* bv  = (const float*)d_in[9];
  const float* Wo  = (const float*)d_in[10];
  const float* bo  = (const float*)d_in[11];
  float* out = (float*)d_out;

  const size_t MD = (size_t)MROWS * DFEAT;   // 4M elems
  const size_t DD = (size_t)DFEAT * DFEAT;   // 1M elems
  half_t* ws  = (half_t*)d_ws;               // 64 MB of f16 scratch total
  half_t* Qh  = ws;
  half_t* Kh  = Qh + MD;
  half_t* Vh  = Kh + MD;
  half_t* Wqh = Vh + MD;
  half_t* Wkh = Wqh + DD;
  half_t* Wvh = Wkh + DD;
  half_t* Woh = Wvh + DD;
  half_t* Qp  = Woh + DD;
  half_t* Kp  = Qp + MD;
  half_t* Vp  = Kp + MD;                     // per-head transposed [B][H][D][S]
  half_t* Cx  = Vp + MD;

  // f32 -> f16 conversions (vectorized: 8 elems / thread / iter)
  cvt_f32_to_f16<<<2048, 256, 0, stream>>>(Q,  Qh,  (int)(MD / 8));
  cvt_f32_to_f16<<<2048, 256, 0, stream>>>(K,  Kh,  (int)(MD / 8));
  cvt_f32_to_f16<<<2048, 256, 0, stream>>>(V,  Vh,  (int)(MD / 8));
  cvt_f32_to_f16<<<512,  256, 0, stream>>>(Wq, Wqh, (int)(DD / 8));
  cvt_f32_to_f16<<<512,  256, 0, stream>>>(Wk, Wkh, (int)(DD / 8));
  cvt_f32_to_f16<<<512,  256, 0, stream>>>(Wv, Wvh, (int)(DD / 8));
  cvt_f32_to_f16<<<512,  256, 0, stream>>>(Wo, Woh, (int)(DD / 8));

  // Q/K/V linear projections (WMMA GEMMs); V written per-head transposed
  dim3 ggrid(DFEAT / 64, MROWS / 64);   // (16, 64)
  gemm_bias<0><<<ggrid, 128, 0, stream>>>(Qh, Wqh, bq, Qp, nullptr);
  gemm_bias<0><<<ggrid, 128, 0, stream>>>(Kh, Wkh, bk, Kp, nullptr);
  gemm_bias<1><<<ggrid, 128, 0, stream>>>(Vh, Wvh, bv, Vp, nullptr);

  // fused masked softmax attention (TDM double-buffered K/V staging)
  attn_fused<<<dim3(SEQ / 64, BATCH * NHEAD), 128, 0, stream>>>(Qp, Kp, Vp, msk, Cx);

  // output projection straight to f32 d_out
  gemm_bias<2><<<ggrid, 128, 0, stream>>>(Cx, Woh, bo, nullptr, out);
}
